// RFGCN_6983616824053
// MI455X (gfx1250) — matrix operand order repaired and checked by reading, not compiled, not run on hardware
//
#include <hip/hip_runtime.h>
#include <stdint.h>

// CDNA5 / gfx1250, wave32. WMMA f32 <- f16 x f16, 16x16x32.
typedef _Float16 v16h __attribute__((ext_vector_type(16)));
typedef float    v8f  __attribute__((ext_vector_type(8)));

#define L_ 4
#define B_ 32
#define S_ 512
#define D_ 64

// LDS layout (bytes):
#define XH_OFF     0          // Xh    : [512][64] f16 = 65536   (f16 copy of gcn_in[b])
#define YINT_OFF   65536      // YinT  : [80][512] f16 = 81920   (Y^T, col 64 = gate, 65..79 pad)
#define YOUTT_OFF  147456     // YoutT : [80][512] f16 = 81920
#define WTIN_OFF   229376     // WTin  : [80][64]  f16 = 10240   (W^T, row 64 = Wg, 65..79 = 0)
#define WTOUT_OFF  239616     // WTout : [80][64]  f16 = 10240
#define BIN_OFF    249856     // biasIn : [80] f32 = 320 (64 = gate bias, 65..79 = 0)
#define BOUT_OFF   250176     // biasOut: [80] f32 = 320
#define ADJS_OFF   250496     // adj scratch: [8 waves][32 rows][16 cols] u8 = 4096
#define SMEM_BYTES 254592

union FragH { v16h h; uint32_t u[8]; uint4 q[2]; };

__device__ __forceinline__ float sigmoidf_(float x) {
    return 1.0f / (1.0f + __expf(-x));
}

// 4 adjacency bytes (0/1) -> two dwords of packed f16 pairs. f16(1.0) = 0x3C00.
__device__ __forceinline__ void cvt4_(uint32_t x, uint32_t& lo, uint32_t& hi) {
    lo = ((x & 0xFFu)         | ((x & 0x0000FF00u) << 8)) * 0x3C00u;
    hi = (((x >> 16) & 0xFFu) | ((x & 0xFF000000u) >> 8)) * 0x3C00u;
}

__global__ void __launch_bounds__(256, 1)
rfgcn_kernel(const float* __restrict__ gcn_in,      // [B,S,D] f32
             const uint8_t* __restrict__ adj,       // [L,B,S,S] bool
             const float* __restrict__ num_words,   // [B]
             const float* __restrict__ W_in,        // [L,D,D]
             const float* __restrict__ b_in,        // [L,D]
             const float* __restrict__ Wg_in,       // [L,D]
             const float* __restrict__ bg_in,       // [L]
             const float* __restrict__ W_out,
             const float* __restrict__ b_out,
             const float* __restrict__ Wg_out,
             const float* __restrict__ bg_out,
             float* __restrict__ out)               // [B,S,D] f32
{
    extern __shared__ char smem[];
    _Float16* Xh     = (_Float16*)(smem + XH_OFF);
    _Float16* YinT   = (_Float16*)(smem + YINT_OFF);
    _Float16* YoutT  = (_Float16*)(smem + YOUTT_OFF);
    _Float16* WTin   = (_Float16*)(smem + WTIN_OFF);
    _Float16* WTout  = (_Float16*)(smem + WTOUT_OFF);
    float*    biasIn  = (float*)(smem + BIN_OFF);
    float*    biasOut = (float*)(smem + BOUT_OFF);

    const int tid    = threadIdx.x;
    const int lane   = tid & 31;
    const int wave   = tid >> 5;
    const int strip  = blockIdx.x;                 // 0..3 (output rows strip*128 .. +127)
    const int b      = blockIdx.y;                 // 0..31
    const int mm     = lane & 15;
    const int hi     = lane >> 4;
    const int m_base = strip * 128 + wave * 16;

    const v8f z8 = {0.f, 0.f, 0.f, 0.f, 0.f, 0.f, 0.f, 0.f};

    // ---- one-time staging: Xh = f16(gcn_in[b]); zero W^T pad rows + bias pads ----
    {
        const float4* gp = (const float4*)(gcn_in + (size_t)b * S_ * D_);
        for (int i = tid; i < (S_ * D_) / 4; i += 256) {
            float4 v = gp[i];
            union { uint2 u2; _Float16 h[4]; } p;
            p.h[0] = (_Float16)v.x; p.h[1] = (_Float16)v.y;
            p.h[2] = (_Float16)v.z; p.h[3] = (_Float16)v.w;
            ((uint2*)Xh)[i] = p.u2;
        }
        uint32_t* w0 = (uint32_t*)(WTin  + 65 * 64);
        uint32_t* w1 = (uint32_t*)(WTout + 65 * 64);
        for (int i = tid; i < (15 * 64) / 2; i += 256) { w0[i] = 0u; w1[i] = 0u; }
        if (tid < 15) { biasIn[65 + tid] = 0.f; biasOut[65 + tid] = 0.f; }
    }

    v8f accFin[4];
    #pragma unroll
    for (int t = 0; t < 4; ++t) accFin[t] = z8;

    for (int l = 0; l < L_; ++l) {
        // ---------- stage W^T (f16) + biases for label l ----------
        for (int idx = tid; idx < 4096; idx += 256) {
            int c = idx >> 6, d = idx & 63;
            WTin [c * 64 + d] = (_Float16)W_in [l * 4096 + d * 64 + c];
            WTout[c * 64 + d] = (_Float16)W_out[l * 4096 + d * 64 + c];
        }
        if (tid < 64) {
            WTin [64 * 64 + tid] = (_Float16)Wg_in [l * 64 + tid];
            WTout[64 * 64 + tid] = (_Float16)Wg_out[l * 64 + tid];
            biasIn [tid] = b_in [l * 64 + tid];
            biasOut[tid] = b_out[l * 64 + tid];
        }
        if (tid == 0) { biasIn[64] = bg_in[l]; biasOut[64] = bg_out[l]; }
        __syncthreads();

        // ---------- phase 1 (WMMA): Y^T[c][j] = f16( (X @ Wext)[j][c] + bias[c] ) ----------
        // M = j (32 tiles, 4 per wave), N = c (5 tiles), K = d (64 -> 2 steps).
        #pragma unroll 1
        for (int it = 0; it < 4; ++it) {
            const int j0 = (wave * 4 + it) * 16;
            FragH aF0, aF1;   // X row fragments for d0 = 0 and d0 = 32
            {
                const uint4* xp = (const uint4*)(Xh + (j0 + mm) * 64 + hi * 8);
                aF0.q[0] = xp[0];   // halves d = hi*8   .. +7
                aF0.q[1] = xp[2];   // halves d = 16+hi*8 .. +7
                aF1.q[0] = xp[4];   // halves d = 32+hi*8 .. +7
                aF1.q[1] = xp[6];   // halves d = 48+hi*8 .. +7
            }
            #pragma unroll
            for (int dir = 0; dir < 2; ++dir) {
                _Float16* WTp = dir ? WTout : WTin;
                const float* bp = dir ? biasOut : biasIn;
                _Float16* Yp  = dir ? YoutT : YinT;
                v8f acc[5];
                #pragma unroll
                for (int t = 0; t < 5; ++t) acc[t] = z8;
                #pragma unroll
                for (int ct = 0; ct < 5; ++ct) {
                    FragH bf;
                    const uint4* wp = (const uint4*)(WTp + (ct * 16 + mm) * 64 + hi * 8);
                    bf.q[0] = wp[0]; bf.q[1] = wp[2];
                    acc[ct] = __builtin_amdgcn_wmma_f32_16x16x32_f16(
                        false, aF0.h, false, bf.h, (short)0, acc[ct], false, false);
                    const uint4* wp2 = (const uint4*)(WTp + (ct * 16 + mm) * 64 + 32 + hi * 8);
                    bf.q[0] = wp2[0]; bf.q[1] = wp2[2];
                    acc[ct] = __builtin_amdgcn_wmma_f32_16x16x32_f16(
                        false, aF1.h, false, bf.h, (short)0, acc[ct], false, false);
                }
                // epilogue: + bias, pack 8 contiguous j (m = hi*8 + r) into one b128 store
                #pragma unroll
                for (int ct = 0; ct < 5; ++ct) {
                    const float bv = bp[ct * 16 + mm];
                    union { uint4 q; _Float16 h[8]; } st;
                    #pragma unroll
                    for (int r = 0; r < 8; ++r) st.h[r] = (_Float16)(acc[ct][r] + bv);
                    *(uint4*)(Yp + (ct * 16 + mm) * 512 + j0 + hi * 8) = st.q;
                }
            }
        }
        __syncthreads();

        // ---------- phase 2 (WMMA): in_t = A^T @ Yin, out_t = A @ Yout ----------
        const uint8_t* adjL = adj + ((size_t)l * B_ + b) * (size_t)S_ * S_;
        uint8_t* adjW = (uint8_t*)(smem + ADJS_OFF) + wave * 512;  // wave-private [32][16]

        v8f accIn[5], accOut[5];
        #pragma unroll
        for (int t = 0; t < 5; ++t) { accIn[t] = z8; accOut[t] = z8; }

        #pragma unroll 1
        for (int k0 = 0; k0 < S_; k0 += 32) {
            // stage transpose panel: row k0+lane, this wave's 16 columns (coalesced b128)
            {
                uint4 rb = *(const uint4*)(adjL + (size_t)(k0 + lane) * S_ + m_base);
                *(uint4*)(adjW + lane * 16) = rb;   // DS in-order per wave: no barrier needed
            }

            FragH aOutF, aInF;
            // A (out-arcs): adj[m_glob, k0+k] -- row-contiguous global bytes
            {
                const uint8_t* rowp = adjL + (size_t)(m_base + mm) * S_ + k0 + hi * 8;
                if (k0 + 32 < S_) __builtin_prefetch(rowp + 32, 0, 1);
                uint32_t r0 = *(const uint32_t*)(rowp);
                uint32_t r1 = *(const uint32_t*)(rowp + 4);
                uint32_t r2 = *(const uint32_t*)(rowp + 16);
                uint32_t r3 = *(const uint32_t*)(rowp + 20);
                cvt4_(r0, aOutF.u[0], aOutF.u[1]);
                cvt4_(r1, aOutF.u[2], aOutF.u[3]);
                cvt4_(r2, aOutF.u[4], aOutF.u[5]);
                cvt4_(r3, aOutF.u[6], aOutF.u[7]);
            }
            // A (in-arcs): adj[k0+k, m_glob] -- transposed read from LDS panel
            {
                #pragma unroll
                for (int v = 0; v < 8; ++v) {
                    const int krel = (v < 4) ? (hi * 8 + 2 * v) : (16 + hi * 8 + 2 * (v - 4));
                    uint32_t b0 = adjW[krel * 16 + mm];
                    uint32_t b1 = adjW[(krel + 1) * 16 + mm];
                    aInF.u[v] = (b0 | (b1 << 16)) * 0x3C00u;
                }
            }

            const int kb = k0 + hi * 8;
            #pragma unroll
            for (int nt = 0; nt < 5; ++nt) {
                const int col = nt * 16 + mm;
                FragH bf;
                const uint4* pI = (const uint4*)(YinT + (size_t)col * S_ + kb);
                bf.q[0] = pI[0]; bf.q[1] = pI[2];
                accIn[nt] = __builtin_amdgcn_wmma_f32_16x16x32_f16(
                    false, aInF.h, false, bf.h, (short)0, accIn[nt], false, false);
                const uint4* pO = (const uint4*)(YoutT + (size_t)col * S_ + kb);
                bf.q[0] = pO[0]; bf.q[1] = pO[2];
                accOut[nt] = __builtin_amdgcn_wmma_f32_16x16x32_f16(
                    false, aOutF.h, false, bf.h, (short)0, accOut[nt], false, false);
            }
        }

        // ---------- gating: gate(row m) = tile4 (m, n=0) -> lanes 0 / 16 ----------
        #pragma unroll
        for (int r = 0; r < 8; ++r) {
            float gi = __shfl(accIn[4][r],  lane & 16, 32);
            float go = __shfl(accOut[4][r], lane & 16, 32);
            float si = sigmoidf_(gi);
            float so = sigmoidf_(go);
            #pragma unroll
            for (int nt = 0; nt < 4; ++nt)
                accFin[nt][r] += accIn[nt][r] * si + accOut[nt][r] * so;
        }
        __syncthreads();   // protect LDS before next label restages
    }

    // ---------- combine: (gcn_in + sum_l acts) / (3 * num_words[b]) ----------
    const float scale = 1.0f / (3.0f * num_words[b]);
    #pragma unroll
    for (int nt = 0; nt < 4; ++nt) {
        #pragma unroll
        for (int r = 0; r < 8; ++r) {
            const int row = m_base + hi * 8 + r;
            const int col = nt * 16 + mm;
            const size_t idx = ((size_t)b * S_ + row) * D_ + col;
            out[idx] = (gcn_in[idx] + accFin[nt][r]) * scale;
        }
    }
}

extern "C" void kernel_launch(void* const* d_in, const int* in_sizes, int n_in,
                              void* d_out, int out_size, void* d_ws, size_t ws_size,
                              hipStream_t stream) {
    const float*   gcn_in    = (const float*)d_in[0];
    const uint8_t* adj       = (const uint8_t*)d_in[1];
    const float*   num_words = (const float*)d_in[2];
    const float*   W_in      = (const float*)d_in[3];
    const float*   b_in      = (const float*)d_in[4];
    const float*   Wg_in     = (const float*)d_in[5];
    const float*   bg_in     = (const float*)d_in[6];
    const float*   W_out     = (const float*)d_in[7];
    const float*   b_out     = (const float*)d_in[8];
    const float*   Wg_out    = (const float*)d_in[9];
    const float*   bg_out    = (const float*)d_in[10];
    float* out = (float*)d_out;

    (void)hipFuncSetAttribute((const void*)rfgcn_kernel,
                              hipFuncAttributeMaxDynamicSharedMemorySize, SMEM_BYTES);

    dim3 grid(4, 32, 1);   // row strips x batches; labels looped internally (no atomics)
    rfgcn_kernel<<<grid, 256, SMEM_BYTES, stream>>>(
        gcn_in, adj, num_words, W_in, b_in, Wg_in, bg_in,
        W_out, b_out, Wg_out, bg_out, out);
}
